// SparseMoEPerTokenFFN_83580063580220
// MI455X (gfx1250) — compile-verified, matrix operand order checked by reading, not run on hardware
//
#include <hip/hip_runtime.h>
#include <math.h>

#define BB 64
#define TT 128
#define EE 4
#define DD 128
#define KK 512

typedef __attribute__((ext_vector_type(16))) _Float16 v16h;
typedef __attribute__((ext_vector_type(2)))  _Float16 h2;
typedef __attribute__((ext_vector_type(4)))  _Float16 h4;
typedef __attribute__((ext_vector_type(8)))  float    v8f;
typedef __attribute__((ext_vector_type(4)))  float    f4;

// Build a 16x32 f16 A/B fragment (per CDNA5 ISA 16-bit A-matrix layout):
// lane L (half h = L>>4): elements 0..7  = K kBase + h*8 + 0..7
//                         elements 8..15 = K kBase + 16 + h*8 + 0..7
__device__ __forceinline__ v16h frag_ld(const _Float16* __restrict__ row, int kBase, int h) {
    v16h r;
    const _Float16* p0 = row + kBase + h * 8;
    const _Float16* p1 = p0 + 16;
#pragma unroll
    for (int i = 0; i < 8; ++i) r[i] = p0[i];
#pragma unroll
    for (int i = 0; i < 8; ++i) r[i + 8] = p1[i];
    return r;
}

// Branch-free exact-erf GELU: Abramowitz-Stegun 7.1.26 (|err_erf| < 1.6e-7).
__device__ __forceinline__ float gelu_erf(float v) {
    float z  = v * 0.70710678118654752f;           // v / sqrt(2)
    float az = __builtin_fabsf(z);
    float t  = __builtin_amdgcn_rcpf(__builtin_fmaf(0.3275911f, az, 1.0f));
    float p  = __builtin_fmaf(t, 1.061405429f, -1.453152027f);
    p = __builtin_fmaf(t, p, 1.421413741f);
    p = __builtin_fmaf(t, p, -0.284496736f);
    p = __builtin_fmaf(t, p, 0.254829592f);
    p = p * t;
    float er = __builtin_fmaf(-p, __expf(-z * z), 1.0f);
    er = __builtin_copysignf(er, z);
    return 0.5f * v * (1.0f + er);
}

// ---------------- router: gates = relu(x @ rw + rb), normalized ----------------
__global__ void moe_router(const float* __restrict__ x, const float* __restrict__ rw,
                           const float* __restrict__ rb, float* __restrict__ gates) {
    int tok = blockIdx.x * blockDim.x + threadIdx.x;  // 0..B*T-1
    if (tok >= BB * TT) return;
    const float* xr = x + (size_t)tok * DD;
    float acc[EE] = {rb[0], rb[1], rb[2], rb[3]};
    for (int d = 0; d < DD; d += 4) {
        f4 xv = *(const f4*)(xr + d);
        const float* wp = rw + d * EE;
#pragma unroll
        for (int e = 0; e < EE; ++e) {
            acc[e] += xv.x * wp[e] + xv.y * wp[EE + e] +
                      xv.z * wp[2 * EE + e] + xv.w * wp[3 * EE + e];
        }
    }
    float s = 1e-9f;
#pragma unroll
    for (int e = 0; e < EE; ++e) { acc[e] = fmaxf(acc[e], 0.0f); s += acc[e]; }
    float inv = 1.0f / s;
#pragma unroll
    for (int e = 0; e < EE; ++e) gates[tok * EE + e] = acc[e] * inv;
}

// ---------------- aux loss: deterministic single-block reduction ----------------
__global__ void moe_aux(const float* __restrict__ gates, float* __restrict__ out_aux) {
    __shared__ float red[256];
    float s = 0.0f;
    for (int i = threadIdx.x; i < BB * TT * EE; i += 256) s += gates[i];
    red[threadIdx.x] = s;
    __syncthreads();
    for (int st = 128; st > 0; st >>= 1) {
        if ((int)threadIdx.x < st) red[threadIdx.x] += red[threadIdx.x + st];
        __syncthreads();
    }
    if (threadIdx.x == 0) out_aux[0] = 0.01f * red[0] / (float)(BB * TT);
}

// ---------------- main per-(t,e) FFN: WMMA f16 pipeline ----------------
__global__ __launch_bounds__(256) void moe_expert_ffn(
    const float* __restrict__ x,  const float* __restrict__ w1g,
    const float* __restrict__ b1g, const float* __restrict__ w2g,
    const float* __restrict__ b2g, float* __restrict__ wsO) {
    __shared__ _Float16 x_lds[BB * DD];   // 16KB  X[b][d]     (f16)
    __shared__ _Float16 h_lds[BB * KK];   // 64KB  H[b][k]     (f16)
    __shared__ _Float16 w_lds[DD * DD];   // 32KB  W^T stage [n][k] (f16)

    const int t = blockIdx.x, e = blockIdx.y;
    const int tid = threadIdx.x;
    const int lane = tid & 31, wv = tid >> 5;
    const int hh = lane >> 4, ln = lane & 15;

    const float* w1 = w1g + (size_t)(t * EE + e) * DD * KK;
    const float* w2 = w2g + (size_t)(t * EE + e) * KK * DD;
    const float* b1 = b1g + (size_t)(t * EE + e) * KK;
    const float* b2 = b2g + (size_t)(t * EE + e) * DD;

    // hoist all per-lane biases into registers (loads issue once, covered by staging)
    float bias1[2][8];
#pragma unroll
    for (int ti = 0; ti < 2; ++ti) {
        int nT = (wv * 2 + ti) >> 2;
#pragma unroll
        for (int c = 0; c < 8; ++c) bias1[ti][c] = b1[c * 64 + nT * 16 + ln];
    }
    float bias2 = b2[wv * 16 + ln];

    // ---- stage X[:,t,:] -> x_lds as f16 (64 rows x 128 cols), b64 packed stores
    {
        int b = tid >> 2, seg = (tid & 3) * 32;
        const float* src = x + ((size_t)b * TT + t) * DD + seg;
        _Float16* dst = x_lds + b * DD + seg;
#pragma unroll
        for (int j = 0; j < 32; j += 4) {
            f4 v = *(const f4*)(src + j);
            h4 hv = {(_Float16)v.x, (_Float16)v.y, (_Float16)v.z, (_Float16)v.w};
            *(h4*)(dst + j) = hv;
        }
    }

    // ================= GEMM1: H = gelu(X[64x128] * W1[128x512] + b1) =================
    // Staging: each thread covers rows (dp, dp+1) x 16 cols; transposed write is a
    // packed half2 -> single ds_store_b32 at even element offset; per-wave dword
    // index = n*64 + dp/2 -> 32 distinct banks (conflict-free).
    const int dp1  = (tid & 63) * 2;       // row pair for W1 stage
    const int seg1 = (tid >> 6) * 16;      // 16 of the 64 chunk columns
    for (int c = 0; c < 8; ++c) {
        __syncthreads();                    // previous chunk's w_lds reads done
        {
            const float* s0 = w1 + (size_t)dp1 * KK + c * 64 + seg1;
            const float* s1 = s0 + KK;
#pragma unroll
            for (int j = 0; j < 16; j += 4) {
                f4 a = *(const f4*)(s0 + j);
                f4 b = *(const f4*)(s1 + j);
                *(h2*)(w_lds + (seg1 + j + 0) * DD + dp1) = h2{(_Float16)a.x, (_Float16)b.x};
                *(h2*)(w_lds + (seg1 + j + 1) * DD + dp1) = h2{(_Float16)a.y, (_Float16)b.y};
                *(h2*)(w_lds + (seg1 + j + 2) * DD + dp1) = h2{(_Float16)a.z, (_Float16)b.z};
                *(h2*)(w_lds + (seg1 + j + 3) * DD + dp1) = h2{(_Float16)a.w, (_Float16)b.w};
            }
            if (c + 1 < 8) __builtin_prefetch(s0 + 64, 0, 1);   // next chunk
        }
        __syncthreads();
        // 16 tiles (4 M x 4 N) per chunk; each wave owns 2
#pragma unroll
        for (int ti = 0; ti < 2; ++ti) {
            int tIdx = wv * 2 + ti;
            int mT = tIdx & 3, nT = tIdx >> 2;
            const _Float16* arow = x_lds + (mT * 16 + ln) * DD;
            const _Float16* brow = w_lds + (nT * 16 + ln) * DD;
            int nG = c * 64 + nT * 16 + ln;          // global hidden index for this lane
            float bias = bias1[ti][c];
            v8f acc;
#pragma unroll
            for (int r = 0; r < 8; ++r) acc[r] = bias;
#pragma unroll
            for (int kk = 0; kk < DD; kk += 32) {
                v16h a  = frag_ld(arow, kk, hh);
                v16h bf = frag_ld(brow, kk, hh);
                acc = __builtin_amdgcn_wmma_f32_16x16x32_f16(
                    false, a, false, bf, (short)0, acc, false, false);
            }
            // branch-free exact GELU, store f16 into h_lds (elem r -> row M=r+8*hh)
#pragma unroll
            for (int r = 0; r < 8; ++r) {
                int M = mT * 16 + r + 8 * hh;
                h_lds[M * KK + nG] = (_Float16)gelu_erf(acc[r]);
            }
        }
    }

    // ================= GEMM2: O = H[64x512] * W2[512x128] + b2 =================
    v8f acc2[4];
#pragma unroll
    for (int i = 0; i < 4; ++i)
#pragma unroll
        for (int r = 0; r < 8; ++r) acc2[i][r] = bias2;

    const int kp2  = (tid & 63) * 2;       // k-row pair for W2 stage
    const int seg2 = (tid >> 6) * 32;      // 32 of the 128 columns
    for (int kc = 0; kc < 4; ++kc) {
        __syncthreads();                    // h_lds writes / prior w_lds reads done
        {   // stage W2[kc*128 + k][dcol] transposed -> w_lds[dcol][k] (packed half2)
            const float* s0 = w2 + (size_t)(kc * DD + kp2) * DD + seg2;
            const float* s1 = s0 + DD;
#pragma unroll
            for (int j = 0; j < 32; j += 4) {
                f4 a = *(const f4*)(s0 + j);
                f4 b = *(const f4*)(s1 + j);
                *(h2*)(w_lds + (seg2 + j + 0) * DD + kp2) = h2{(_Float16)a.x, (_Float16)b.x};
                *(h2*)(w_lds + (seg2 + j + 1) * DD + kp2) = h2{(_Float16)a.y, (_Float16)b.y};
                *(h2*)(w_lds + (seg2 + j + 2) * DD + kp2) = h2{(_Float16)a.z, (_Float16)b.z};
                *(h2*)(w_lds + (seg2 + j + 3) * DD + kp2) = h2{(_Float16)a.w, (_Float16)b.w};
            }
            if (kc + 1 < 4) __builtin_prefetch(s0 + DD * DD, 0, 1);
        }
        __syncthreads();
#pragma unroll
        for (int i = 0; i < 4; ++i) {       // mTile = i, nTile = wv
            const _Float16* arow = h_lds + (i * 16 + ln) * KK + kc * DD;
            const _Float16* brow = w_lds + (wv * 16 + ln) * DD;
#pragma unroll
            for (int kk = 0; kk < DD; kk += 32) {
                v16h a  = frag_ld(arow, kk, hh);
                v16h bf = frag_ld(brow, kk, hh);
                acc2[i] = __builtin_amdgcn_wmma_f32_16x16x32_f16(
                    false, a, false, bf, (short)0, acc2[i], false, false);
            }
        }
    }

    // store per-expert output O[t][e][b][d] (ungated) to workspace
    float* obase = wsO + (size_t)(t * EE + e) * BB * DD;
#pragma unroll
    for (int i = 0; i < 4; ++i) {
#pragma unroll
        for (int r = 0; r < 8; ++r) {
            int b = i * 16 + r + 8 * hh;
            int d = wv * 16 + ln;
            obase[b * DD + d] = acc2[i][r];
        }
    }
}

// ---------------- combine: out[b,t,:] = sum_e gates[b,t,e] * O[t,e,b,:] ----------------
__global__ void moe_combine(const float* __restrict__ wsO, const float* __restrict__ gates,
                            float* __restrict__ out) {
    int gid = blockIdx.x * blockDim.x + threadIdx.x;   // one float4 each
    const int nTot = BB * TT * DD / 4;
    if (gid >= nTot) return;
    int base = gid * 4;
    int b = base / (TT * DD);
    int rem = base % (TT * DD);
    int t = rem / DD;
    int d = rem % DD;
    const float* g = gates + (b * TT + t) * EE;
    f4 acc = {0.0f, 0.0f, 0.0f, 0.0f};
#pragma unroll
    for (int e = 0; e < EE; ++e) {
        f4 v = *(const f4*)(wsO + ((size_t)(t * EE + e) * BB + b) * DD + d);
        acc += v * g[e];
    }
    *(f4*)(out + base) = acc;
}

extern "C" void kernel_launch(void* const* d_in, const int* in_sizes, int n_in,
                              void* d_out, int out_size, void* d_ws, size_t ws_size,
                              hipStream_t stream) {
    (void)in_sizes; (void)n_in; (void)out_size; (void)ws_size;
    const float* x  = (const float*)d_in[0];
    const float* w1 = (const float*)d_in[1];
    const float* b1 = (const float*)d_in[2];
    const float* w2 = (const float*)d_in[3];
    const float* b2 = (const float*)d_in[4];
    const float* rw = (const float*)d_in[5];
    const float* rb = (const float*)d_in[6];
    float* out   = (float*)d_out;
    float* gates = (float*)d_ws;                              // B*T*E f32 (128KB)
    float* wsO   = gates + (size_t)BB * TT * EE;              // T*E*B*D f32 (16MB)

    moe_router<<<dim3((BB * TT + 255) / 256), dim3(256), 0, stream>>>(x, rw, rb, gates);
    moe_aux<<<dim3(1), dim3(256), 0, stream>>>(gates, out + (size_t)BB * TT * DD);
    moe_expert_ffn<<<dim3(TT, EE), dim3(256), 0, stream>>>(x, w1, b1, w2, b2, wsO);
    moe_combine<<<dim3((BB * TT * DD / 4 + 255) / 256), dim3(256), 0, stream>>>(wsO, gates, out);
}
